// SparseConvolutionBase_83769042141676
// MI455X (gfx1250) — matrix operand order repaired and checked by reading, not compiled, not run on hardware
//
#include <hip/hip_runtime.h>

#define K_VOL   27
#define PAIRS   60000
#define NPTS    100000
#define C       64
#define TILES   (PAIRS / 16)      // 3750 tiles of 16 pairs, exact
#define WAVES   8
#define BLOCK   256
#define BLOCKS_X 59               // ~8 tiles per wave -> resident-B amortized

typedef __attribute__((ext_vector_type(16))) __bf16 v16bf;
typedef __attribute__((ext_vector_type(8)))  __bf16 v8bf;
typedef __attribute__((ext_vector_type(4)))  __bf16 v4bf;
typedef __attribute__((ext_vector_type(8)))  float  v8f;

#define WMMA_BF16(A, B, ACC) \
    __builtin_amdgcn_wmma_f32_16x16x32_bf16(false, (A), false, (B), (short)0, (ACC), false, false)

// ---------------------------------------------------------------------------
// out[n][c] = bias[c]   (vectorized float4 writes; runs before the atomics)
// ---------------------------------------------------------------------------
__global__ void bias_init_kernel(const float* __restrict__ bias,
                                 float* __restrict__ out) {
    int i = blockIdx.x * blockDim.x + threadIdx.x;     // index over float4s
    if (i < NPTS * C / 4) {
        float4 b = *(const float4*)(bias + ((i * 4) & (C - 1)));
        ((float4*)out)[i] = b;
    }
}

// ---------------------------------------------------------------------------
// Split f32 input into hi/lo bf16 arrays (x ~= hi + lo, ~f32 accuracy).
// ---------------------------------------------------------------------------
__global__ void prep_input_kernel(const float* __restrict__ in,
                                  __bf16* __restrict__ ihi,
                                  __bf16* __restrict__ ilo) {
    int i = blockIdx.x * blockDim.x + threadIdx.x;     // index over float4s
    if (i < NPTS * C / 4) {
        float4 x = ((const float4*)in)[i];
        float xs[4] = { x.x, x.y, x.z, x.w };
        v4bf h, l;
        #pragma unroll
        for (int e = 0; e < 4; ++e) {
            __bf16 hb = (__bf16)xs[e];
            h[e] = hb;
            l[e] = (__bf16)(xs[e] - (float)hb);
        }
        ((v4bf*)ihi)[i] = h;
        ((v4bf*)ilo)[i] = l;
    }
}

// ---------------------------------------------------------------------------
// Pre-lay-out kernel weights as hi/lo bf16 WMMA B-fragments:
//   Bpre[k][s][nt][lane][e],  e=0..15, lane=0..31, nt=0..3, s=0..1
// B layout (16-bit, 32x16): lane holds col N = nt*16 + lane%16,
//                           K = s*32 + (lane>=16)*8 + (e&7) + (e>=8)*16.
// ---------------------------------------------------------------------------
__global__ void prep_weight_kernel(const float* __restrict__ w,
                                   __bf16* __restrict__ bhi,
                                   __bf16* __restrict__ blo) {
    const int k = blockIdx.x;
    for (int i = threadIdx.x; i < 4096; i += 256) {
        const int e    = i & 15;
        const int lane = (i >> 4) & 31;
        const int nt   = (i >> 9) & 3;
        const int s    = (i >> 11) & 1;
        const int kk   = s * 32 + (lane >> 4) * 8 + (e & 7) + (e >> 3) * 16;   // Cin (K) index
        const int col  = nt * 16 + (lane & 15);           // Cout (N) index
        float  f = w[k * C * C + kk * C + col];
        __bf16 h = (__bf16)f;
        bhi[k * 4096 + i] = h;
        blo[k * 4096 + i] = (__bf16)(f - (float)h);
    }
}

// ---------------------------------------------------------------------------
// Fast path: everything pre-converted to bf16 hi/lo in WMMA-native layouts.
// grid = (BLOCKS_X, K_VOL), block = 256 (8 waves). Each wave: grid-stride
// over 16-pair tiles; gather bf16 rows -> 24 WMMAs -> f32 atomic scatter.
// ---------------------------------------------------------------------------
__global__ __launch_bounds__(BLOCK) void sparse_conv_wmma_pre(
    const __bf16* __restrict__ ihi,     // [NPTS, C] hi
    const __bf16* __restrict__ ilo,     // [NPTS, C] lo
    const __bf16* __restrict__ bhi,     // [K_VOL, 4096] fragment order
    const __bf16* __restrict__ blo,     // [K_VOL, 4096] fragment order
    const int*    __restrict__ in_map,  // [K_VOL, PAIRS]
    const int*    __restrict__ out_map, // [K_VOL, PAIRS]
    float*        __restrict__ out)     // [NPTS, C]
{
    const int k    = blockIdx.y;
    const int lane = threadIdx.x & 31;
    const int wave = threadIdx.x >> 5;
    const int half = lane >> 4;
    const int l16  = lane & 15;

    // ---- resident B fragments: coalesced 32 B/lane loads, zero VALU --------
    v16bf bH[2][4], bL[2][4];
    #pragma unroll
    for (int s = 0; s < 2; ++s)
        #pragma unroll
        for (int nt = 0; nt < 4; ++nt) {
            const int off = k * 4096 + ((s * 4 + nt) * 32 + lane) * 16;
            bH[s][nt] = *(const v16bf*)(bhi + off);
            bL[s][nt] = *(const v16bf*)(blo + off);
        }

    const int* imap   = in_map  + k * PAIRS;
    const int* omap   = out_map + k * PAIRS;
    const int  stride = gridDim.x * WAVES;
    const int  tile0  = blockIdx.x * WAVES + wave;

    // Row index for the first tile; rotated forward each iteration.
    int arow = (tile0 < TILES) ? imap[tile0 * 16 + l16] : 0;

    for (int tile = tile0; tile < TILES; tile += stride) {
        const int p0 = tile * 16;

        // ---- issue ALL independent loads up front (covered by WMMA chain) --
        int4 o0 = *(const int4*)(omap + p0 + half * 8);
        int4 o1 = *(const int4*)(omap + p0 + half * 8 + 4);

        const __bf16* hrow = ihi + arow * C;   // i32 offset -> saddr+voffset
        const __bf16* lrow = ilo + arow * C;
        // A layout (16-bit, 16x32): K groups [b..b+7],[b+16..b+23], b=s*32+half*8
        const int b0 = half * 8;               // k-step 0
        v8bf h00 = *(const v8bf*)(hrow + b0);
        v8bf h01 = *(const v8bf*)(hrow + b0 + 16);
        v8bf l00 = *(const v8bf*)(lrow + b0);
        v8bf l01 = *(const v8bf*)(lrow + b0 + 16);
        const int b1 = 32 + half * 8;          // k-step 1
        v8bf h10 = *(const v8bf*)(hrow + b1);
        v8bf h11 = *(const v8bf*)(hrow + b1 + 16);
        v8bf l10 = *(const v8bf*)(lrow + b1);
        v8bf l11 = *(const v8bf*)(lrow + b1 + 16);

        // ---- next-tile row index + prefetches (L2-resident working set) ----
        int nrow = arow;
        if (tile + stride < TILES) {
            nrow = imap[(tile + stride) * 16 + l16];
            __builtin_prefetch(ihi + nrow * C, 0, 0);
            __builtin_prefetch(ilo + nrow * C, 0, 0);
            __builtin_prefetch(omap + (tile + stride) * 16, 0, 0);
        }

        v16bf ah[2], al[2];
        ah[0] = __builtin_shufflevector(h00, h01, 0,1,2,3,4,5,6,7,8,9,10,11,12,13,14,15);
        al[0] = __builtin_shufflevector(l00, l01, 0,1,2,3,4,5,6,7,8,9,10,11,12,13,14,15);
        ah[1] = __builtin_shufflevector(h10, h11, 0,1,2,3,4,5,6,7,8,9,10,11,12,13,14,15);
        al[1] = __builtin_shufflevector(l10, l11, 0,1,2,3,4,5,6,7,8,9,10,11,12,13,14,15);

        v8f acc[4] = {};
        #pragma unroll
        for (int s = 0; s < 2; ++s)
            #pragma unroll
            for (int nt = 0; nt < 4; ++nt) {
                acc[nt] = WMMA_BF16(ah[s], bH[s][nt], acc[nt]);   // hi*hi
                acc[nt] = WMMA_BF16(ah[s], bL[s][nt], acc[nt]);   // hi*lo
                acc[nt] = WMMA_BF16(al[s], bH[s][nt], acc[nt]);   // lo*hi
            }

        // ---- scatter-add: C layout r <-> M = r + half*8, N = l16 ------------
        const int orows[8] = { o0.x, o0.y, o0.z, o0.w, o1.x, o1.y, o1.z, o1.w };
        #pragma unroll
        for (int r = 0; r < 8; ++r) {
            float* op = out + (orows[r] * C + l16);   // i32 offset addressing
            #pragma unroll
            for (int nt = 0; nt < 4; ++nt)
                atomicAdd(op + nt * 16, acc[nt][r]);
        }

        arow = nrow;
    }
}

// ---------------------------------------------------------------------------
// Fallback path (ws too small): convert on the fly, weights staged in LDS.
// ---------------------------------------------------------------------------
__global__ __launch_bounds__(BLOCK) void sparse_conv_wmma_fly(
    const float* __restrict__ input,
    const float* __restrict__ weight,
    const int*   __restrict__ in_map,
    const int*   __restrict__ out_map,
    float*       __restrict__ out)
{
    __shared__ float ldsW[C * C];

    const int k    = blockIdx.y;
    const int tid  = threadIdx.x;
    const int lane = tid & 31;
    const int wave = tid >> 5;
    const int half = lane >> 4;
    const int l16  = lane & 15;

    {   // exact 4 float4s per thread: no guards, no exec-mask branches
        const float4* wsrc = (const float4*)(weight + k * C * C);
        float4*       wdst = (float4*)ldsW;
        #pragma unroll
        for (int j = 0; j < (C * C / 4) / BLOCK; ++j)
            wdst[tid + j * BLOCK] = wsrc[tid + j * BLOCK];
    }
    __syncthreads();

    v16bf bH[2][4], bL[2][4];
    #pragma unroll
    for (int s = 0; s < 2; ++s) {
        const int kbase = s * 32 + half * 8;
        #pragma unroll
        for (int nt = 0; nt < 4; ++nt) {
            const int col = nt * 16 + l16;
            #pragma unroll
            for (int e = 0; e < 16; ++e) {
                float  f = ldsW[(kbase + (e & 7) + (e >> 3) * 16) * C + col];
                __bf16 h = (__bf16)f;
                bH[s][nt][e] = h;
                bL[s][nt][e] = (__bf16)(f - (float)h);
            }
        }
    }

    const int* imap   = in_map  + k * PAIRS;
    const int* omap   = out_map + k * PAIRS;
    const int  stride = gridDim.x * WAVES;

    for (int tile = blockIdx.x * WAVES + wave; tile < TILES; tile += stride) {
        const int p0   = tile * 16;
        const int arow = imap[p0 + l16];
        const float* src = input + arow * C;          // i32 offset addressing

        int4 o0 = *(const int4*)(omap + p0 + half * 8);
        int4 o1 = *(const int4*)(omap + p0 + half * 8 + 4);

        v8f acc[4] = {};
        #pragma unroll
        for (int s = 0; s < 2; ++s) {
            const int b = s * 32 + half * 8;
            float4 x0 = *(const float4*)(src + b);
            float4 x1 = *(const float4*)(src + b + 4);
            float4 y0 = *(const float4*)(src + b + 16);
            float4 y1 = *(const float4*)(src + b + 20);
            float xs[16] = { x0.x, x0.y, x0.z, x0.w,  x1.x, x1.y, x1.z, x1.w,
                             y0.x, y0.y, y0.z, y0.w,  y1.x, y1.y, y1.z, y1.w };
            v16bf ah, al;
            #pragma unroll
            for (int e = 0; e < 16; ++e) {
                __bf16 h = (__bf16)xs[e];
                ah[e] = h;
                al[e] = (__bf16)(xs[e] - (float)h);
            }
            #pragma unroll
            for (int nt = 0; nt < 4; ++nt) {
                acc[nt] = WMMA_BF16(ah, bH[s][nt], acc[nt]);
                acc[nt] = WMMA_BF16(ah, bL[s][nt], acc[nt]);
                acc[nt] = WMMA_BF16(al, bH[s][nt], acc[nt]);
            }
        }

        const int orows[8] = { o0.x, o0.y, o0.z, o0.w, o1.x, o1.y, o1.z, o1.w };
        #pragma unroll
        for (int r = 0; r < 8; ++r) {
            float* op = out + (orows[r] * C + l16);
            #pragma unroll
            for (int nt = 0; nt < 4; ++nt)
                atomicAdd(op + nt * 16, acc[nt][r]);
        }
    }
}

// ---------------------------------------------------------------------------
extern "C" void kernel_launch(void* const* d_in, const int* in_sizes, int n_in,
                              void* d_out, int out_size, void* d_ws, size_t ws_size,
                              hipStream_t stream) {
    const float* input  = (const float*)d_in[0];   // [NPTS, C] f32
    const float* weight = (const float*)d_in[1];   // [K_VOL, C, C] f32
    const float* bias   = (const float*)d_in[2];   // [1, C] f32
    const int*   in_map = (const int*)d_in[3];     // [K_VOL, PAIRS] i32
    const int*   out_map= (const int*)d_in[4];     // [K_VOL, PAIRS] i32
    float*       out    = (float*)d_out;           // [NPTS, C] f32

    const int n4 = NPTS * C / 4;
    bias_init_kernel<<<(n4 + 255) / 256, 256, 0, stream>>>(bias, out);

    const size_t szIn = (size_t)NPTS * C * sizeof(__bf16);     // 12.8 MB each
    const size_t szB  = (size_t)K_VOL * 4096 * sizeof(__bf16); // 216 KB each
    const size_t need = 2 * szIn + 2 * szB;

    dim3 grid(BLOCKS_X, K_VOL);
    if (ws_size >= need) {
        __bf16* ihi = (__bf16*)d_ws;
        __bf16* ilo = (__bf16*)((char*)d_ws + szIn);
        __bf16* bhi = (__bf16*)((char*)d_ws + 2 * szIn);
        __bf16* blo = (__bf16*)((char*)d_ws + 2 * szIn + szB);

        prep_input_kernel<<<(n4 + 255) / 256, 256, 0, stream>>>(input, ihi, ilo);
        prep_weight_kernel<<<K_VOL, 256, 0, stream>>>(weight, bhi, blo);
        sparse_conv_wmma_pre<<<grid, BLOCK, 0, stream>>>(ihi, ilo, bhi, blo,
                                                         in_map, out_map, out);
    } else {
        sparse_conv_wmma_fly<<<grid, BLOCK, 0, stream>>>(input, weight,
                                                         in_map, out_map, out);
    }
}
